// CausalSelfAttention_2903397892582
// MI455X (gfx1250) — matrix operand order repaired and checked by reading, not compiled, not run on hardware
//
#include <hip/hip_runtime.h>
#include <math.h>

// ---------------------------------------------------------------------------
// CDNA5 (gfx1250) CausalSelfAttention + DPP-gated attention.
// All GEMM-shaped work uses v_wmma_f32_16x16x32_bf16 (wave32 WMMA).
// B-side operands are pre-packed into fragment-ready order so every WMMA
// fragment is a contiguous 32B (2x b128) load; V is stored transposed for
// the P@V stage for the same reason.
// ---------------------------------------------------------------------------

typedef __attribute__((ext_vector_type(16))) __bf16 v16bf;
typedef __attribute__((ext_vector_type(8)))  __bf16 v8bf;
typedef __attribute__((ext_vector_type(8)))  float  v8f;

#define T_SEQ   512
#define C_DIM   768
#define NHEAD   12
#define HD      64
#define BATCH   2
#define MROWS   (BATCH * T_SEQ)   // 1024
#define N_QKV   (3 * C_DIM)       // 2304
#define NEG_INF (-1e30f)
#define EPSV    (1e-6f)

static __device__ __forceinline__ v8f wmma_bf16(v16bf a, v16bf b, v8f c) {
    return __builtin_amdgcn_wmma_f32_16x16x32_bf16(false, a, false, b,
                                                   (short)0, c, false, false);
}
static __device__ __forceinline__ v16bf cat16(v8bf lo, v8bf hi) {
    return __builtin_shufflevector(lo, hi, 0, 1, 2, 3, 4, 5, 6, 7,
                                   8, 9, 10, 11, 12, 13, 14, 15);
}
// A-fragment (16x32 bf16) for a row-major [M,K] source: lane l, row M=l&15,
// needs K = [16g..16g+15]  (g = l>>4) -> two v8bf loads.
static __device__ __forceinline__ v16bf load_a_frag(const __bf16* row, int g) {
    const v8bf lo = *(const v8bf*)(row + 16 * g);
    const v8bf hi = *(const v8bf*)(row + 16 * g + 8);
    return cat16(lo, hi);
}

// ---------------------------------------------------------------------------
// f32 -> bf16 conversion (grid-stride)
// ---------------------------------------------------------------------------
__global__ void cvt_bf16_kernel(const float* __restrict__ src,
                                __bf16* __restrict__ dst, int n) {
    for (int idx = blockIdx.x * blockDim.x + threadIdx.x; idx < n;
         idx += gridDim.x * blockDim.x)
        dst[idx] = (__bf16)src[idx];
}

// ---------------------------------------------------------------------------
// Fused f32->bf16 conversion + pack of a row-major [K,N] weight into WMMA
// B-fragment order: Wpk[((kt*(N/16)+nt)*32 + lane)*16 + e], where element
// e=2v+u of lane (g=lane>>4, ml=lane&15) is W[kt*32 + 2v + 16g + u][nt*16+ml].
// ---------------------------------------------------------------------------
__global__ void pack_w_kernel(const float* __restrict__ W,
                              __bf16* __restrict__ Wpk, int K, int N) {
    const int total = K * N;
    for (int idx = blockIdx.x * blockDim.x + threadIdx.x; idx < total;
         idx += gridDim.x * blockDim.x) {
        const int e    = idx & 15;
        const int lane = (idx >> 4) & 31;
        const int tile = idx >> 9;
        const int ntn  = N >> 4;
        const int kt = tile / ntn, nt = tile % ntn;
        const int g = lane >> 4, ml = lane & 15;
        const int v = e >> 1, u = e & 1;
        const int kk  = kt * 32 + 2 * v + 16 * g + u;
        const int col = nt * 16 + ml;
        Wpk[idx] = (__bf16)W[(size_t)kk * N + col];
    }
}

// ---------------------------------------------------------------------------
// QKV GEMM: [1024,768]bf16 x packed[768,2304]bf16 + bias, fused head-scatter
// epilogue writing q/k/v f32 head-major, q/k bf16 head-major, v bf16
// head-major TRANSPOSED ([head][dim][T]) for the attention P@V stage.
// One wave per 16x16 output tile.
// ---------------------------------------------------------------------------
__global__ __launch_bounds__(128) void gemm_qkv_kernel(
    const __bf16* __restrict__ Abf, const v16bf* __restrict__ Bpk,
    const float* __restrict__ bias,
    float* __restrict__ qf, float* __restrict__ kf, float* __restrict__ vf,
    __bf16* __restrict__ qb, __bf16* __restrict__ kb,
    __bf16* __restrict__ vbt) {
    const int lane = threadIdx.x & 31;
    const int wave = threadIdx.x >> 5;
    const int tile = blockIdx.x * (blockDim.x >> 5) + wave;
    const int ntn  = N_QKV / 16;  // 144
    const int tm = tile / ntn, tn = tile % ntn;
    if (tm >= MROWS / 16) return;
    const int m0 = tm * 16, n0 = tn * 16;
    const int g = lane >> 4, ml = lane & 15;

    const __bf16* arow = Abf + (size_t)(m0 + ml) * C_DIM;
    v8f acc = {0.f, 0.f, 0.f, 0.f, 0.f, 0.f, 0.f, 0.f};
    for (int k0 = 0, kt = 0; k0 < C_DIM; k0 += 32, ++kt) {
        const v16bf a = load_a_frag(arow + k0, g);
        const v16bf b = Bpk[((size_t)kt * ntn + tn) * 32 + lane];
        __builtin_prefetch(arow + k0 + 32, 0, 1);
        __builtin_prefetch(&Bpk[((size_t)(kt + 1) * ntn + tn) * 32 + lane], 0, 1);
        acc = wmma_bf16(a, b, acc);
    }

    // Epilogue: column -> (which of q/k/v, head, dim); row -> (batch, token).
    const int col   = n0 + ml;
    const int which = col / C_DIM;
    const int inner = col % C_DIM;
    const int h = inner / HD, d = inner % HD;
    const float bc = bias[col];
    float*  fdst = (which == 0) ? qf : (which == 1) ? kf : vf;
    __bf16* bdst = (which == 0) ? qb : kb;  // unused when which==2
#pragma unroll
    for (int r = 0; r < 8; ++r) {
        const int m = m0 + r + 8 * g;
        const int bidx = m / T_SEQ, t = m % T_SEQ;
        const float val = acc[r] + bc;
        const size_t bh = (size_t)(bidx * NHEAD + h);
        fdst[(bh * T_SEQ + t) * HD + d] = val;
        if (which == 2)
            vbt[(bh * HD + d) * T_SEQ + t] = (__bf16)val;  // transposed
        else
            bdst[(bh * T_SEQ + t) * HD + d] = (__bf16)val;
    }
}

// ---------------------------------------------------------------------------
// Standard causal attention, one wave per (head, 16-query tile).
// Pass 1: S = QK^T/8 via WMMA into LDS, track row max/sum (flash-style sums).
// Pass 2: O = exp(S-max) @ V via WMMA, exp computed on-the-fly in A-frag load;
// V fragments are contiguous vector loads from the transposed V buffer.
// ---------------------------------------------------------------------------
__global__ __launch_bounds__(32) void attn_std_kernel(
    const __bf16* __restrict__ qb, const __bf16* __restrict__ kb,
    const __bf16* __restrict__ vbt, float* __restrict__ ystd) {
    __shared__ float Sbuf[16][T_SEQ + 8];
    __shared__ float lmaxsh[16];

    const int lane = threadIdx.x;
    const int head = blockIdx.x / (T_SEQ / 16);
    const int qt   = blockIdx.x % (T_SEQ / 16);
    const int m0   = qt * 16;
    const __bf16* qh = qb + (size_t)head * T_SEQ * HD;
    const __bf16* kh = kb + (size_t)head * T_SEQ * HD;
    const __bf16* vT = vbt + (size_t)head * HD * T_SEQ;
    const int g = lane >> 4, ml = lane & 15;

    // Q A-fragments for K-dims [0,32) and [32,64)
    const __bf16* qrow = qh + (size_t)(m0 + ml) * HD;
    const v16bf aq0 = load_a_frag(qrow, g);
    const v16bf aq1 = load_a_frag(qrow + 32, g);

    float rmax[8], rsum[8];
#pragma unroll
    for (int r = 0; r < 8; ++r) { rmax[r] = NEG_INF; rsum[r] = 0.f; }
    const float scale = 0.125f;  // 1/sqrt(64)

    for (int j0 = 0; j0 <= m0; j0 += 16) {
        // B fragment: B[kk][n] = k[j0+n][kk]; per lane (n=ml) dims are the
        // contiguous chunk [16g,16g+16) (bk0) and [32+16g,32+16g+16) (bk1).
        const v16bf* krow = (const v16bf*)(kh + (size_t)(j0 + ml) * HD);
        const v16bf bk0 = krow[g];
        const v16bf bk1 = krow[2 + g];
        v8f s = {0.f, 0.f, 0.f, 0.f, 0.f, 0.f, 0.f, 0.f};
        s = wmma_bf16(aq0, bk0, s);
        s = wmma_bf16(aq1, bk1, s);
#pragma unroll
        for (int r = 0; r < 8; ++r) {
            const int mrow = r + 8 * g;  // C-layout row in tile
            float sv = s[r] * scale;
            if (j0 == m0 && ml > mrow) sv = NEG_INF;  // causal mask (diag tile)
            Sbuf[mrow][j0 + ml] = sv;
            float tm = sv;  // width-16 max reduction
#pragma unroll
            for (int off = 8; off >= 1; off >>= 1)
                tm = fmaxf(tm, __shfl_xor(tm, off, 32));
            const float nm = fmaxf(rmax[r], tm);
            float ts = __expf(sv - nm);
#pragma unroll
            for (int off = 8; off >= 1; off >>= 1)
                ts += __shfl_xor(ts, off, 32);
            rsum[r] = rsum[r] * __expf(rmax[r] - nm) + ts;
            rmax[r] = nm;
        }
    }
    if (lane == 0)
        for (int r = 0; r < 8; ++r) lmaxsh[r] = rmax[r];
    if (lane == 16)
        for (int r = 0; r < 8; ++r) lmaxsh[8 + r] = rmax[r];
    __syncthreads();

    // Pass 2: O accumulation, 32 keys per step, 4 N-chunks of 16 dims.
    // Max key touched = 480+31 = 511, so no bounds clamp needed on V.
    v8f o0 = {0.f,0.f,0.f,0.f,0.f,0.f,0.f,0.f}, o1 = o0, o2 = o0, o3 = o0;
    for (int j0 = 0; j0 < m0 + 16; j0 += 32) {
        v16bf ap;  // P fragment, exp on the fly (A row = ml)
#pragma unroll
        for (int v = 0; v < 8; ++v) {
            const int ka = 2 * v + 16 * g;
#pragma unroll
            for (int u = 0; u < 2; ++u) {
                const int key = j0 + ka + u;
                float p = 0.f;
                if (key < m0 + 16) p = __expf(Sbuf[ml][key] - lmaxsh[ml]);
                ap[2 * v + u] = (__bf16)p;
            }
        }
        // V B-fragments: dim row = c*16+ml in transposed V, keys contiguous.
        const int kchunk = (j0 >> 4) + g;
        const v16bf bv0 = ((const v16bf*)(vT + (size_t)(0 * 16 + ml) * T_SEQ))[kchunk];
        const v16bf bv1 = ((const v16bf*)(vT + (size_t)(1 * 16 + ml) * T_SEQ))[kchunk];
        const v16bf bv2 = ((const v16bf*)(vT + (size_t)(2 * 16 + ml) * T_SEQ))[kchunk];
        const v16bf bv3 = ((const v16bf*)(vT + (size_t)(3 * 16 + ml) * T_SEQ))[kchunk];
        o0 = wmma_bf16(ap, bv0, o0);
        o1 = wmma_bf16(ap, bv1, o1);
        o2 = wmma_bf16(ap, bv2, o2);
        o3 = wmma_bf16(ap, bv3, o3);
    }
#pragma unroll
    for (int r = 0; r < 8; ++r) {
        const float inv = 1.f / rsum[r];
        const int row = m0 + r + 8 * g;
        float* yrow = ystd + ((size_t)head * T_SEQ + row) * HD;
        yrow[0 * 16 + ml] = o0[r] * inv;
        yrow[1 * 16 + ml] = o1[r] * inv;
        yrow[2 * 16 + ml] = o2[r] * inv;
        yrow[3 * 16 + ml] = o3[r] * inv;
    }
}

// ---------------------------------------------------------------------------
// DPP greedy-selection attention: one wave32 per (batch*head, token).
// top-16 causal candidates by k.q similarity, fast greedy-MAP via incremental
// Cholesky (bordered-Gram Schur complement == det ratio), then softmax
// attention over the selected subset (f32 throughout for det stability).
// ---------------------------------------------------------------------------
__global__ __launch_bounds__(32) void dpp_attn_kernel(
    const float* __restrict__ qf, const float* __restrict__ kf,
    const float* __restrict__ vf, float* __restrict__ ydpp) {
    __shared__ float qsh[HD];
    __shared__ int   topid[17];
    __shared__ float gram[17][18];
    __shared__ int   selid[8];

    const int lane = threadIdx.x;
    const int bh = blockIdx.x / T_SEQ;
    const int i  = blockIdx.x % T_SEQ;
    const float* kh = kf + (size_t)bh * T_SEQ * HD;
    const float* vh = vf + (size_t)bh * T_SEQ * HD;
    const float* qi = qf + ((size_t)bh * T_SEQ + i) * HD;

    qsh[lane]      = qi[lane];
    qsh[lane + 32] = qi[lane + 32];
    __syncthreads();

    // causal similarities; lane owns tokens [lane*16, lane*16+16)
    float sv[16];
#pragma unroll
    for (int j = 0; j < 16; ++j) {
        const int t = lane * 16 + j;
        float s = NEG_INF;
        if (t <= i) {
            float acc = 0.f;
            for (int d = 0; d < HD; ++d) acc += kh[t * HD + d] * qsh[d];
            s = acc;
        }
        sv[j] = s;
    }

    // top-16 by value (ties -> lower index), iterative argmax
    for (int sel = 0; sel < 16; ++sel) {
        float bvv = -3.0e38f;
        int   bt  = 1 << 30;
#pragma unroll
        for (int j = 0; j < 16; ++j) {
            const int t = lane * 16 + j;
            if (sv[j] > bvv || (sv[j] == bvv && t < bt)) { bvv = sv[j]; bt = t; }
        }
#pragma unroll
        for (int off = 16; off >= 1; off >>= 1) {
            const float ov = __shfl_xor(bvv, off, 32);
            const int   ot = __shfl_xor(bt, off, 32);
            if (ov > bvv || (ov == bvv && ot < bt)) { bvv = ov; bt = ot; }
        }
        if (lane == 0) topid[sel] = bt;
        if ((bt >> 4) == lane) sv[bt & 15] = -3.4e38f;  // mask picked
    }
    if (lane == 0) { topid[16] = i; selid[0] = i; }
    __syncthreads();

    // 17x17 Gram of candidate keys (+ key_i as item 16)
    if (lane < 17) {
        const int tc = topid[lane];
        for (int jj = 0; jj < 17; ++jj) {
            const int tj = topid[jj];
            float acc = 0.f;
            for (int d = 0; d < HD; ++d) acc += kh[tc * HD + d] * kh[tj * HD + d];
            gram[lane][jj] = acc;
        }
    }
    __syncthreads();

    const int limit = (i + 1 < 16) ? (i + 1) : 16;
    unsigned actmask;
    {
        const bool mine = (lane < 16) && (lane < limit) && (topid[lane] != i);
        actmask = (unsigned)(__ballot(mine) & 0xFFFFull);
    }

    // fast greedy MAP: lane c owns candidate c's Cholesky row z[] + residual d2
    float z[7];
    float d2 = (lane < 17) ? gram[lane][lane] : 1.f;
    int   selcount = 1;
    float proddet  = gram[16][16];  // det of {k_i}
    float cur      = logf(proddet + EPSV);
    int   lastsel  = 16;
    bool  stopped  = false;

    for (int t = 0; t < 7; ++t) {
        // Cholesky border update vs most recently selected item
        const float d2b = __shfl(d2, lastsel, 32);
        float dot = 0.f;
#pragma unroll
        for (int idx = 0; idx < 7; ++idx) {
            const float zb = __shfl(z[idx], lastsel, 32);
            if (idx < t) dot += z[idx] * zb;
        }
        const float gcb = (lane < 17) ? gram[lane][lastsel] : 0.f;
        const float e = (gcb - dot) / sqrtf(fmaxf(d2b, 1e-30f));
        z[t] = e;
        d2 -= e * e;

        // candidate scores: log(det(sel + cand) + eps) / size
        const float detc  = proddet * d2;
        const float score = logf(detc + EPSV) / (float)(selcount + 1);

        float bvv = -3.0e38f;
        int   bi  = 1 << 30;
        if (lane < 16 && ((actmask >> lane) & 1u)) { bvv = score; bi = lane; }
#pragma unroll
        for (int off = 16; off >= 1; off >>= 1) {
            const float ov = __shfl_xor(bvv, off, 32);
            const int   oi = __shfl_xor(bi, off, 32);
            if (ov > bvv || (ov == bvv && oi < bi)) { bvv = ov; bi = oi; }
        }
        const bool any  = (actmask != 0u);
        const bool take = any && !stopped && ((bvv > cur) || (selcount < 4));
        if (take) {
            if (lane == 0) selid[selcount] = topid[bi];
            cur      = bvv;
            proddet  = __shfl(detc, bi, 32);  // proddet * d2_b
            actmask &= ~(1u << bi);
            lastsel  = bi;
            selcount += 1;
        } else {
            stopped = true;
        }
    }
    __syncthreads();

    // attention over selected subset
    float logit = NEG_INF;
    if (lane < selcount) {
        const int tok = selid[lane];
        float acc = 0.f;
        for (int d = 0; d < HD; ++d) acc += kh[tok * HD + d] * qsh[d];
        logit = acc * 0.125f;
    }
    float mx = logit;
#pragma unroll
    for (int off = 4; off >= 1; off >>= 1)
        mx = fmaxf(mx, __shfl_xor(mx, off, 32));
    float ex = (lane < selcount) ? __expf(logit - mx) : 0.f;
    float sm = ex;
#pragma unroll
    for (int off = 4; off >= 1; off >>= 1) sm += __shfl_xor(sm, off, 32);
    const float smtot = __shfl(sm, 0, 32);

    float out0 = 0.f, out1 = 0.f;
    for (int s = 0; s < 8; ++s) {
        if (s < selcount) {
            const float w = __shfl(ex, s, 32) / smtot;
            const int tok = selid[s];
            out0 += w * vh[tok * HD + lane];
            out1 += w * vh[tok * HD + lane + 32];
        }
    }
    float* yd = ydpp + ((size_t)bh * T_SEQ + i) * HD;
    yd[lane]      = out0;
    yd[lane + 32] = out1;
}

// ---------------------------------------------------------------------------
// gate combine: y = sigmoid(g)*y_dpp + (1-sigmoid(g))*y_std, head-major ->
// [B,T,C] bf16 for the projection GEMM.
// ---------------------------------------------------------------------------
__global__ void combine_gate_kernel(const float* __restrict__ ystd,
                                    const float* __restrict__ ydpp,
                                    const float* __restrict__ gatep,
                                    __bf16* __restrict__ ycomb, int n) {
    const int idx = blockIdx.x * blockDim.x + threadIdx.x;
    if (idx >= n) return;
    const float gate = 1.f / (1.f + __expf(-gatep[0]));
    const int bh  = idx / (T_SEQ * HD);
    const int rem = idx % (T_SEQ * HD);
    const int t = rem / HD, d = rem % HD;
    const int b = bh / NHEAD, h = bh % NHEAD;
    const float y = gate * ydpp[idx] + (1.f - gate) * ystd[idx];
    ycomb[((size_t)(b * T_SEQ + t)) * C_DIM + h * HD + d] = (__bf16)y;
}

// ---------------------------------------------------------------------------
// Projection GEMM: [1024,768]bf16 x packed[768,768]bf16 + bias -> d_out f32
// ---------------------------------------------------------------------------
__global__ __launch_bounds__(128) void gemm_proj_kernel(
    const __bf16* __restrict__ Abf, const v16bf* __restrict__ Bpk,
    const float* __restrict__ bias, float* __restrict__ Out) {
    const int lane = threadIdx.x & 31;
    const int wave = threadIdx.x >> 5;
    const int tile = blockIdx.x * (blockDim.x >> 5) + wave;
    const int ntn  = C_DIM / 16;  // 48
    const int tm = tile / ntn, tn = tile % ntn;
    if (tm >= MROWS / 16) return;
    const int m0 = tm * 16, n0 = tn * 16;
    const int g = lane >> 4, ml = lane & 15;

    const __bf16* arow = Abf + (size_t)(m0 + ml) * C_DIM;
    v8f acc = {0.f, 0.f, 0.f, 0.f, 0.f, 0.f, 0.f, 0.f};
    for (int k0 = 0, kt = 0; k0 < C_DIM; k0 += 32, ++kt) {
        const v16bf a = load_a_frag(arow + k0, g);
        const v16bf b = Bpk[((size_t)kt * ntn + tn) * 32 + lane];
        __builtin_prefetch(arow + k0 + 32, 0, 1);
        __builtin_prefetch(&Bpk[((size_t)(kt + 1) * ntn + tn) * 32 + lane], 0, 1);
        acc = wmma_bf16(a, b, acc);
    }
    const float bc = bias[n0 + ml];
#pragma unroll
    for (int r = 0; r < 8; ++r)
        Out[(size_t)(m0 + r + 8 * g) * C_DIM + n0 + ml] = acc[r] + bc;
}

// ---------------------------------------------------------------------------
extern "C" void kernel_launch(void* const* d_in, const int* in_sizes, int n_in,
                              void* d_out, int out_size, void* d_ws,
                              size_t ws_size, hipStream_t stream) {
    const float* x      = (const float*)d_in[0];  // [2,512,768]
    const float* w_qkv  = (const float*)d_in[1];  // [768,2304]
    const float* b_qkv  = (const float*)d_in[2];  // [2304]
    const float* w_proj = (const float*)d_in[3];  // [768,768]
    const float* b_proj = (const float*)d_in[4];  // [768]
    const float* gatep  = (const float*)d_in[5];  // [1]
    float* out = (float*)d_out;                   // [2,512,768]

    // workspace carve-up (256B aligned)
    char* base = (char*)d_ws;
    size_t off = 0;
    auto carve = [&](size_t bytes) {
        void* p = base + off;
        off = (off + bytes + 255) & ~(size_t)255;
        return p;
    };
    const size_t nX  = (size_t)MROWS * C_DIM;               // 786432
    const size_t nWq = (size_t)C_DIM * N_QKV;               // 1769472
    const size_t nWp = (size_t)C_DIM * C_DIM;               // 589824
    const size_t nH  = (size_t)BATCH * NHEAD * T_SEQ * HD;  // 786432

    __bf16* xb     = (__bf16*)carve(nX * 2);
    __bf16* wqkvp  = (__bf16*)carve(nWq * 2);   // fragment-packed
    __bf16* wprojp = (__bf16*)carve(nWp * 2);   // fragment-packed
    float*  qf  = (float*)carve(nH * 4);
    float*  kf  = (float*)carve(nH * 4);
    float*  vf  = (float*)carve(nH * 4);
    __bf16* qb  = (__bf16*)carve(nH * 2);
    __bf16* kb  = (__bf16*)carve(nH * 2);
    __bf16* vbt = (__bf16*)carve(nH * 2);       // transposed [head][dim][T]
    float*  ystd  = (float*)carve(nH * 4);
    float*  ydpp  = (float*)carve(nH * 4);
    __bf16* ycomb = (__bf16*)carve(nX * 2);
    (void)ws_size; (void)in_sizes; (void)n_in; (void)out_size;

    // 1) convert x to bf16; convert+pack weights into fragment order
    cvt_bf16_kernel<<<(int)((nX + 255) / 256), 256, 0, stream>>>(x, xb, (int)nX);
    pack_w_kernel<<<(int)((nWq + 255) / 256), 256, 0, stream>>>(w_qkv, wqkvp,
                                                                C_DIM, N_QKV);
    pack_w_kernel<<<(int)((nWp + 255) / 256), 256, 0, stream>>>(w_proj, wprojp,
                                                                C_DIM, C_DIM);

    // 2) QKV GEMM + head scatter: (1024/16)*(2304/16) = 9216 tiles, 4 waves/blk
    gemm_qkv_kernel<<<9216 / 4, 128, 0, stream>>>(
        xb, (const v16bf*)wqkvp, b_qkv, qf, kf, vf, qb, kb, vbt);

    // 3) standard causal attention: 24 heads * 32 query tiles, 1 wave each
    attn_std_kernel<<<BATCH * NHEAD * (T_SEQ / 16), 32, 0, stream>>>(qb, kb, vbt,
                                                                     ystd);

    // 4) DPP greedy attention: one wave per (head, token)
    dpp_attn_kernel<<<BATCH * NHEAD * T_SEQ, 32, 0, stream>>>(qf, kf, vf, ydpp);

    // 5) gated combine -> [B,T,C] bf16
    combine_gate_kernel<<<(int)((nH + 255) / 256), 256, 0, stream>>>(
        ystd, ydpp, gatep, ycomb, (int)nH);

    // 6) output projection -> d_out (f32)
    gemm_proj_kernel<<<(64 * 48) / 4, 128, 0, stream>>>(
        ycomb, (const v16bf*)wprojp, b_proj, out);
}
